// GIN_72232759984565
// MI455X (gfx1250) — matrix operand order repaired and checked by reading, not compiled, not run on hardware
//
#include <hip/hip_runtime.h>
#include <hip/hip_bf16.h>

#define N_NODES 8192
#define D 64
#define ROWS_PER_BLK 128
#define KCHUNK 32
#define NCHUNKS (N_NODES / KCHUNK)
#define XP_CHUNK_U32 1024            /* one packed X chunk = 64 cols x 16 dwords */

typedef __attribute__((ext_vector_type(16))) __bf16 v16bf;
typedef __attribute__((ext_vector_type(8)))  float  v8f;
typedef __attribute__((ext_vector_type(4)))  float  v4f;
typedef __attribute__((ext_vector_type(4)))  unsigned int v4u;
typedef int v4i_vec __attribute__((vector_size(16)));   // matches builtin's V4i

#if defined(__has_builtin)
# if __has_builtin(__builtin_amdgcn_global_load_async_to_lds_b128)
#  define HAVE_ASYNC_LDS 1
# endif
# if __has_builtin(__builtin_amdgcn_s_wait_asynccnt)
#  define HAVE_WAIT_ASYNC 1
# endif
#endif
#ifndef HAVE_ASYNC_LDS
# define HAVE_ASYNC_LDS 0
#endif
#ifndef HAVE_WAIT_ASYNC
# define HAVE_WAIT_ASYNC 0
#endif

static __device__ __forceinline__ unsigned short bf_bits(float f) {
    __bf16 h = (__bf16)f;
    unsigned short u;
    __builtin_memcpy(&u, &h, 2);
    return u;
}
static __device__ __forceinline__ unsigned int pack_bf2(float lo, float hi) {
    return (unsigned int)bf_bits(lo) | ((unsigned int)bf_bits(hi) << 16);
}
static __device__ __forceinline__ __bf16 bits_bf(unsigned short u) {
    __bf16 h;
    __builtin_memcpy(&h, &u, 2);
    return h;
}

// 16B global -> LDS copy; async (ASYNCcnt) when the toolchain exposes the builtin.
static __device__ __forceinline__ void copy16_g2l(const unsigned int* gsrc,
                                                  unsigned int* ldst) {
#if HAVE_ASYNC_LDS
    __builtin_amdgcn_global_load_async_to_lds_b128(
        (__attribute__((address_space(1))) v4i_vec*)gsrc,
        (__attribute__((address_space(3))) v4i_vec*)ldst,
        0, 0);
#else
    v4u v = *(const v4u*)gsrc;
    *(v4u*)ldst = v;
#endif
}
static __device__ __forceinline__ void wait_async0() {
#if HAVE_ASYNC_LDS
# if HAVE_WAIT_ASYNC
    __builtin_amdgcn_s_wait_asynccnt(0);
# else
    asm volatile("s_wait_asynccnt 0x0" ::: "memory");
# endif
#endif
}

// Pre-pack X (f32 [N][64]) into per-chunk bf16-pair tiles:
// Xp[chunk][c][j] = pack(X[32*chunk+2j][c], X[32*chunk+2j+1][c]), j<16, c<64.
__global__ __launch_bounds__(256)
void pack_x_kernel(const float* __restrict__ X, unsigned int* __restrict__ Xp)
{
    int u = blockIdx.x * 256 + threadIdx.x;   // 0..262143
    int c = u & 63;                           // coalesced along c
    int j = (u >> 6) & 15;
    int chunk = u >> 10;
    int k = chunk * KCHUNK + 2 * j;
    unsigned int v = pack_bf2(X[(size_t)k * D + c], X[(size_t)(k + 1) * D + c]);
    Xp[(size_t)chunk * XP_CHUNK_U32 + (size_t)c * 16 + j] = v;
}

// Hout = relu( ((A + I) @ X) @ W + bias ); also emits HpOut = packed-bf16 form
// of Hout in the Xp tile layout (consumed as next layer's B operand).
__global__ __launch_bounds__(256)
void gin_layer_kernel(const float* __restrict__ A,
                      const float* __restrict__ X,          // f32, for self-loop
                      const unsigned int* __restrict__ Xp,  // packed bf16 pairs
                      const float* __restrict__ W,
                      const float* __restrict__ bias,
                      float* __restrict__ Hout,
                      unsigned int* __restrict__ HpOut)
{
    __shared__ unsigned int   tileB[2][D][20];   // X chunk, pad 20 dw (16B-aligned rows)
    __shared__ unsigned int   tileW[D][36];      // W packed pairs, pad 36 dw
    __shared__ unsigned short sS[8][16][66];     // per-wave S strip, bf16 row-major

    const int tid  = threadIdx.x;
    const int wave = tid >> 5;
    const int lane = tid & 31;
    const int nn   = lane & 15;
    const int hi   = lane >> 4;
    const int row0 = blockIdx.x * ROWS_PER_BLK + wave * 16;
    const int myrow = row0 + nn;
    const int base2 = hi ? 8 : 0;    // A-fragment K sub-offset
    const int jb    = hi ? 4 : 0;    // B-fragment dword start
    const int fc    = tid >> 2;      // fill: column 0..63
    const int fg    = tid & 3;       // fill: dword group 0..3

    // ---- stage W into LDS (once) ----
    for (int i = 0; i < 8; ++i) {
        int e = tid + 256 * i;
        int c = e & 63;
        int j = e >> 6;
        tileW[c][j] = pack_bf2(W[(2 * j) * D + c], W[(2 * j + 1) * D + c]);
    }
    // ---- prime chunk 0 (one contiguous 16B per thread, coalesced) ----
    copy16_g2l(Xp + 4 * tid, &tileB[0][fc][4 * fg]);
    wait_async0();
    __syncthreads();

    v8f acc[4];
    for (int t = 0; t < 4; ++t)
        for (int i = 0; i < 8; ++i) acc[t][i] = 0.0f;

    const float* Arow = A + (size_t)myrow * N_NODES;

    // ================= K loop over the 8192-wide adjacency =================
    for (int kc = 0; kc < NCHUNKS; ++kc) {
        const int cur = kc & 1;
        const int kb  = kc * KCHUNK;

        if (kc + 1 < NCHUNKS) {
            copy16_g2l(Xp + (size_t)(kc + 1) * XP_CHUNK_U32 + 4 * tid,
                       &tileB[cur ^ 1][fc][4 * fg]);
            __builtin_prefetch(Arow + kb + KCHUNK + base2, 0, 0);
        }

        // ---- A fragment: 16 fp32 of this lane's row (streamed NT), -> bf16 ----
        const v4f* ap  = (const v4f*)(Arow + kb + base2);
        const v4f* ap2 = (const v4f*)(Arow + kb + base2 + 16);
        v4f f0 = __builtin_nontemporal_load(ap);
        v4f f1 = __builtin_nontemporal_load(ap + 1);
        v4f f2 = __builtin_nontemporal_load(ap2);
        v4f f3 = __builtin_nontemporal_load(ap2 + 1);
        v16bf a;
        a[0]  = (__bf16)f0.x; a[1]  = (__bf16)f0.y; a[2]  = (__bf16)f0.z; a[3]  = (__bf16)f0.w;
        a[4]  = (__bf16)f1.x; a[5]  = (__bf16)f1.y; a[6]  = (__bf16)f1.z; a[7]  = (__bf16)f1.w;
        a[8]  = (__bf16)f2.x; a[9]  = (__bf16)f2.y; a[10] = (__bf16)f2.z; a[11] = (__bf16)f2.w;
        a[12] = (__bf16)f3.x; a[13] = (__bf16)f3.y; a[14] = (__bf16)f3.z; a[15] = (__bf16)f3.w;

        // ---- 4 N-tiles of B from LDS, WMMA accumulate ----
        for (int t = 0; t < 4; ++t) {
            const v4u* bp = (const v4u*)&tileB[cur][16 * t + nn][jb];
            v16bf bfr;
            ((v4u*)&bfr)[0] = bp[0];
            ((v4u*)&bfr)[1] = bp[2];
            acc[t] = __builtin_amdgcn_wmma_f32_16x16x32_bf16(
                false, a, false, bfr, (short)0, acc[t], false, false);
        }
        wait_async0();
        __syncthreads();
    }

    // ---- self-loop: (A+I)@X = A@X + X ----
    for (int t = 0; t < 4; ++t)
        for (int i = 0; i < 8; ++i) {
            int r = row0 + i + hi * 8;
            acc[t][i] += X[(size_t)r * D + 16 * t + nn];
        }

    // ---- spill S strip to LDS as bf16 for A-fragment reload ----
    for (int t = 0; t < 4; ++t)
        for (int i = 0; i < 8; ++i)
            sS[wave][i + hi * 8][16 * t + nn] = bf_bits(acc[t][i]);
    __syncthreads();

    // ================= fused 64x64 weight GEMM: H = relu(S @ W + b) =================
    v8f acc2[4];
    for (int t = 0; t < 4; ++t)
        for (int i = 0; i < 8; ++i) acc2[t][i] = 0.0f;

    const unsigned short* srow = &sS[wave][nn][0];
    for (int q = 0; q < 2; ++q) {
        v16bf s;
        for (int g = 0; g < 2; ++g) {
            int kk = 32 * q + base2 + 16 * g;
            for (int p = 0; p < 4; ++p) {
                unsigned int v = *(const unsigned int*)&srow[kk + 2 * p];
                s[g * 8 + 2 * p]     = bits_bf((unsigned short)(v & 0xffffu));
                s[g * 8 + 2 * p + 1] = bits_bf((unsigned short)(v >> 16));
            }
        }
        const int j0 = 16 * q + jb;
        for (int t = 0; t < 4; ++t) {
            const v4u* bp = (const v4u*)&tileW[16 * t + nn][j0];
            v16bf wfr;
            ((v4u*)&wfr)[0] = bp[0];
            ((v4u*)&wfr)[1] = bp[2];
            acc2[t] = __builtin_amdgcn_wmma_f32_16x16x32_bf16(
                false, s, false, wfr, (short)0, acc2[t], false, false);
        }
    }

    // ---- bias + relu; store f32 and packed-bf16 (next layer's B operand) ----
    for (int t = 0; t < 4; ++t) {
        float bc = bias[16 * t + nn];
        int c = 16 * t + nn;
        for (int p = 0; p < 4; ++p) {
            float v0 = fmaxf(acc2[t][2 * p]     + bc, 0.0f);
            float v1 = fmaxf(acc2[t][2 * p + 1] + bc, 0.0f);
            int r = row0 + hi * 8 + 2 * p;        // even row of the pair
            Hout[(size_t)r * D + c]       = v0;
            Hout[(size_t)(r + 1) * D + c] = v1;
            HpOut[(size_t)(r >> 5) * XP_CHUNK_U32 + (size_t)c * 16 + ((r & 31) >> 1)]
                = pack_bf2(v0, v1);
        }
    }
}

// rs-pool (power iteration) + classifier + log_softmax. One block per graph.
__global__ __launch_bounds__(128)
void gin_pool_kernel(const float* __restrict__ H,
                     const float* __restrict__ Wl,
                     const float* __restrict__ bl,
                     float* __restrict__ out)
{
    __shared__ float vec[64];
    __shared__ float ssum;
    __shared__ float outv[10];
    __shared__ float lse;

    const int g = blockIdx.x, tid = threadIdx.x;
    float h[64];
    const float* hrow = H + ((size_t)g * 128 + tid) * (size_t)D;
    for (int c = 0; c < 64; ++c) h[c] = hrow[c];

    if (tid < 64) vec[tid] = 0.0f;
    __syncthreads();
    for (int k = 0; k < 64; ++k) {
        int c = (tid + k) & 63;
        atomicAdd(&vec[c], h[c]);
    }
    __syncthreads();
    if (tid == 0) {
        float s = 0.0f;
        for (int c = 0; c < 64; ++c) s += vec[c] * vec[c];
        ssum = 1.0f / (sqrtf(s) + 1e-8f);
    }
    __syncthreads();
    if (tid < 64) vec[tid] *= ssum;
    __syncthreads();

    for (int it = 0; it < 5; ++it) {
        float y = 0.0f;
        for (int c = 0; c < 64; ++c) y += h[c] * vec[c];
        __syncthreads();
        if (tid < 64) vec[tid] = 0.0f;
        __syncthreads();
        for (int k = 0; k < 64; ++k) {
            int c = (tid + k) & 63;
            atomicAdd(&vec[c], y * h[c]);
        }
        __syncthreads();
        if (tid == 0) {
            float s = 0.0f;
            for (int c = 0; c < 64; ++c) s += vec[c] * vec[c];
            ssum = 1.0f / (sqrtf(s) + 1e-8f);
        }
        __syncthreads();
        if (tid < 64) vec[tid] *= ssum;
        __syncthreads();
    }

    float y = 0.0f;
    for (int c = 0; c < 64; ++c) y += h[c] * vec[c];
    if (tid == 0) ssum = 0.0f;
    __syncthreads();
    atomicAdd(&ssum, y * y);
    __syncthreads();

    if (tid < 10) {
        float sv = sqrtf(ssum);
        float o = bl[tid];
        for (int c = 0; c < 64; ++c) o += vec[c] * sv * Wl[c * 10 + tid];
        outv[tid] = o;
    }
    __syncthreads();
    if (tid == 0) {
        float m = -1e30f;
        for (int j = 0; j < 10; ++j) m = fmaxf(m, outv[j]);
        float s = 0.0f;
        for (int j = 0; j < 10; ++j) s += expf(outv[j] - m);
        lse = m + logf(s);
    }
    __syncthreads();
    if (tid < 10) out[g * 10 + tid] = outv[tid] - lse;
}

extern "C" void kernel_launch(void* const* d_in, const int* in_sizes, int n_in,
                              void* d_out, int out_size, void* d_ws, size_t ws_size,
                              hipStream_t stream)
{
    const float* x_in = (const float*)d_in[0];
    const float* adj  = (const float*)d_in[1];
    // d_in[2] = idx: implied by layout (node / 128)
    const float* W1 = (const float*)d_in[3];
    const float* b1 = (const float*)d_in[4];
    const float* W2 = (const float*)d_in[5];
    const float* b2 = (const float*)d_in[6];
    const float* Wl = (const float*)d_in[7];
    const float* bl = (const float*)d_in[8];
    float* out = (float*)d_out;

    const size_t XP_U32 = (size_t)NCHUNKS * XP_CHUNK_U32;   // 262144 dwords = 1 MB
    unsigned int* Xp0 = (unsigned int*)d_ws;
    float*        H1  = (float*)(Xp0 + XP_U32);             // 2 MB
    unsigned int* Hp1 = (unsigned int*)(H1 + (size_t)N_NODES * D);
    float*        H2  = (float*)(Hp1 + XP_U32);             // 2 MB
    unsigned int* Hp2 = (unsigned int*)(H2 + (size_t)N_NODES * D);  // written, unused

    pack_x_kernel<<<(N_NODES / 2) * D / 256, 256, 0, stream>>>(x_in, Xp0);
    gin_layer_kernel<<<N_NODES / ROWS_PER_BLK, 256, 0, stream>>>(adj, x_in, Xp0, W1, b1, H1, Hp1);
    gin_layer_kernel<<<N_NODES / ROWS_PER_BLK, 256, 0, stream>>>(adj, H1, Hp1, W2, b2, H2, Hp2);
    gin_pool_kernel<<<64, 128, 0, stream>>>(H2, Wl, bl, out);
}
